// GraphSAGE_64476049047967
// MI455X (gfx1250) — compile-verified, matrix operand order checked by reading, not run on hardware
//
#include <hip/hip_runtime.h>
#include <hip/hip_bf16.h>

#define NNODES 100000
#define NEDGES 3200000
#define DDIM 256
#define KTOT 512

typedef __attribute__((ext_vector_type(16))) __bf16 v16bf;
typedef __attribute__((ext_vector_type(8)))  float  v8f;

// A-fragment (16x32 bf16) per-lane K index for element e (0..15), half = lane/16.
// ISA 7.12.2: lanes 0-15: VGPR0..3 -> K=0..7, VGPR4..7 -> K=16..23
//             lanes 16-31: VGPR0..3 -> K=8..15, VGPR4..7 -> K=24..31
// Equivalent: e=0..7 -> k = 8*half + e ; e=8..15 -> k = 16 + 8*half + (e-8)
__device__ __forceinline__ int amap(int e, int half) {
    return (e < 8) ? (8 * half + e) : (16 + 8 * half + (e - 8));
}

// ---------------- utility kernels ----------------

__global__ void zero_f32(float* __restrict__ p, long long n) {
    long long i = (long long)blockIdx.x * blockDim.x + threadIdx.x;
    if (i < n) p[i] = 0.0f;
}

__global__ void deg_kernel(const long long* __restrict__ tgt, float* __restrict__ deg, int nE) {
    int e = blockIdx.x * blockDim.x + threadIdx.x;
    if (e < nE) atomicAdd(&deg[(int)tgt[e]], 1.0f);
}

__global__ void invdeg_kernel(float* __restrict__ deg, int n) {
    int i = blockIdx.x * blockDim.x + threadIdx.x;
    if (i < n) {
        float d = deg[i];
        deg[i] = 1.0f / fmaxf(d, 1.0f);
    }
}

// Pre-swizzle concat([W_l; W_r]) (512x256 f32) into B-fragment order, bf16:
// Wsw[(((kb*16 + colTile)*32 + lane)*16 + e] = Wcat[(kb*32 + amap(e,half))*256 + colTile*16 + lane%16]
// so each (kb, colTile, lane) fragment is 16 contiguous bf16 = 32 bytes.
__global__ void swizzle_w(const float* __restrict__ Wl, const float* __restrict__ Wr,
                          __bf16* __restrict__ Wsw) {
    int i = blockIdx.x * blockDim.x + threadIdx.x;   // 16*16*32*16 = 131072
    if (i >= 16 * 16 * 32 * 16) return;
    int e    = i & 15;
    int lane = (i >> 4) & 31;
    int ct   = (i >> 9) & 15;
    int kb   = i >> 13;
    int half = lane >> 4, lm = lane & 15;
    int k = kb * 32 + amap(e, half);
    int n = ct * 16 + lm;
    float v = (k < DDIM) ? Wl[k * DDIM + n] : Wr[(k - DDIM) * DDIM + n];
    Wsw[i] = (__bf16)v;
}

// ---------------- edge scatter: agg[tgt] += x[src] ----------------
// 64 threads per edge, 4 channels (float4 gather) per thread.
__global__ void __launch_bounds__(256)
scatter_add(const long long* __restrict__ src, const long long* __restrict__ tgt,
            const float* __restrict__ xin, float* __restrict__ agg, int nE) {
    long long idx = (long long)blockIdx.x * blockDim.x + threadIdx.x;
    int e = (int)(idx >> 6);
    if (e >= nE) return;
    int c = (int)(idx & 63) << 2;
    int s = (int)src[e];
    int t = (int)tgt[e];
    const float4 v = *(const float4*)(xin + (long long)s * DDIM + c);
    float* p = agg + (long long)t * DDIM + c;
    atomicAdd(p + 0, v.x);
    atomicAdd(p + 1, v.y);
    atomicAdd(p + 2, v.z);
    atomicAdd(p + 3, v.w);
}

// ---------------- fused WMMA GEMM ----------------
// out[i,:] = act( (agg[i,:]*invdeg[i]) @ W_l + x[i,:] @ W_r + b )
//          = act( concat(agg*inv, x)[i,:] @ Wcat + b ),  K = 512
// One wave computes a 16x64 tile; A loaded as 4x b128 (two contiguous 8-float
// runs per lane), B fragments pre-swizzled so each is 2x b128.
__global__ void __launch_bounds__(256)
sage_gemm(const float* __restrict__ agg, const float* __restrict__ invdeg,
          const float* __restrict__ xin, const __bf16* __restrict__ Wsw,
          const float* __restrict__ bias, float* __restrict__ out, int do_relu) {
    const int waveInBlk = threadIdx.x >> 5;
    const int lane = threadIdx.x & 31;
    const int half = lane >> 4;
    const int lm = lane & 15;

    // 6250 row tiles x 4 col groups = 25000 wave jobs; 8 waves / block
    const int waveId = blockIdx.x * 8 + waveInBlk;
    const int rowTile = waveId >> 2;       // 0..6249
    const int colGrp  = waveId & 3;        // 0..3 (64 cols each)

    const int row = rowTile * 16 + lm;     // A-matrix row for this lane
    const float sc = invdeg[row];
    const float* aggRow = agg + (long long)row * DDIM;
    const float* xRow   = xin + (long long)row * DDIM;

    const v16bf* wfrag = (const v16bf*)Wsw;   // indexed [(kb*16 + ct)*32 + lane]

    v8f acc0 = {}, acc1 = {}, acc2 = {}, acc3 = {};

    #pragma unroll 2
    for (int kb = 0; kb < 16; ++kb) {      // K = 512, 32 per step
        const bool isAgg = (kb < 8);
        const float* arow = isAgg ? aggRow : xRow;
        const int kloc = (isAgg ? kb : (kb - 8)) * 32 + 8 * half;
        const float s = isAgg ? sc : 1.0f;

        // A: two contiguous 8-float runs -> 4 x global_load_b128
        const float4* pa = (const float4*)(arow + kloc);
        float4 f0 = pa[0], f1 = pa[1], f2 = pa[4], f3 = pa[5]; // +0,+16B, +64f? no: see below
        // pa[0],pa[1] cover k = kloc..kloc+7 ; runs are kloc and kloc+16 floats
        f2 = *((const float4*)(arow + kloc + 16));
        f3 = *((const float4*)(arow + kloc + 20));

        v16bf afrag;
        #pragma unroll
        for (int q = 0; q < 4; ++q) {
            float4 f = (q == 0) ? f0 : (q == 1) ? f1 : (q == 2) ? f2 : f3;
            afrag[q * 4 + 0] = (__bf16)(f.x * s);
            afrag[q * 4 + 1] = (__bf16)(f.y * s);
            afrag[q * 4 + 2] = (__bf16)(f.z * s);
            afrag[q * 4 + 3] = (__bf16)(f.w * s);
        }

        // B: 4 pre-swizzled fragments, each 32 contiguous bytes per lane
        const int fbase = (kb * 16 + colGrp * 4) * 32 + lane;
        v16bf b0 = wfrag[fbase];
        v16bf b1 = wfrag[fbase + 32];
        v16bf b2 = wfrag[fbase + 64];
        v16bf b3 = wfrag[fbase + 96];

        acc0 = __builtin_amdgcn_wmma_f32_16x16x32_bf16(false, afrag, false, b0,
                                                       (short)0, acc0, false, false);
        acc1 = __builtin_amdgcn_wmma_f32_16x16x32_bf16(false, afrag, false, b1,
                                                       (short)0, acc1, false, false);
        acc2 = __builtin_amdgcn_wmma_f32_16x16x32_bf16(false, afrag, false, b2,
                                                       (short)0, acc2, false, false);
        acc3 = __builtin_amdgcn_wmma_f32_16x16x32_bf16(false, afrag, false, b3,
                                                       (short)0, acc3, false, false);
    }

    // C/D layout: VGPR v, lane -> m = v + 8*half, n = lm
    #pragma unroll
    for (int c = 0; c < 4; ++c) {
        v8f acc = (c == 0) ? acc0 : (c == 1) ? acc1 : (c == 2) ? acc2 : acc3;
        const int col = colGrp * 64 + c * 16 + lm;
        const float bb = bias[col];
        #pragma unroll
        for (int v = 0; v < 8; ++v) {
            const int m = v + 8 * half;
            float r = acc[v] + bb;
            if (do_relu) r = fmaxf(r, 0.0f);
            out[(long long)(rowTile * 16 + m) * DDIM + col] = r;
        }
    }
}

// ---------------- host side ----------------

extern "C" void kernel_launch(void* const* d_in, const int* in_sizes, int n_in,
                              void* d_out, int out_size, void* d_ws, size_t ws_size,
                              hipStream_t stream) {
    const float*     x     = (const float*)d_in[0];
    const long long* edges = (const long long*)d_in[1];  // int64, shape (2, E)
    const float*     W_l1  = (const float*)d_in[2];
    const float*     b1    = (const float*)d_in[3];
    const float*     W_r1  = (const float*)d_in[4];
    const float*     W_l2  = (const float*)d_in[5];
    const float*     b2    = (const float*)d_in[6];
    const float*     W_r2  = (const float*)d_in[7];
    float* out = (float*)d_out;

    const long long* src = edges;
    const long long* tgt = edges + NEDGES;

    // workspace carve-out (256B aligned)
    char* ws = (char*)d_ws;
    const long long featBytes = (long long)NNODES * DDIM * sizeof(float); // 102.4 MB
    float*  agg  = (float*)ws;  ws += featBytes;
    float*  h    = (float*)ws;  ws += featBytes;
    float*  deg  = (float*)ws;  ws += ((long long)NNODES * 4 + 255) / 256 * 256;
    __bf16* Wsw1 = (__bf16*)ws; ws += (long long)KTOT * DDIM * 2;
    __bf16* Wsw2 = (__bf16*)ws;

    const long long nFeat = (long long)NNODES * DDIM;

    // degree (once; same edges both layers), then invert in place
    zero_f32<<<(NNODES + 255) / 256, 256, 0, stream>>>(deg, NNODES);
    deg_kernel<<<(NEDGES + 255) / 256, 256, 0, stream>>>(tgt, deg, NEDGES);
    invdeg_kernel<<<(NNODES + 255) / 256, 256, 0, stream>>>(deg, NNODES);

    // pre-swizzled bf16 weights (fragment order)
    swizzle_w<<<(KTOT * DDIM + 255) / 256, 256, 0, stream>>>(W_l1, W_r1, Wsw1);
    swizzle_w<<<(KTOT * DDIM + 255) / 256, 256, 0, stream>>>(W_l2, W_r2, Wsw2);

    const long long scatterThreads = (long long)NEDGES * 64;
    const int scatterBlocks = (int)((scatterThreads + 255) / 256);
    const int gemmBlocks = (NNODES / 16) * 4 / 8;   // 6250 row tiles * 4 colgrps / 8 waves

    // ---- layer 1 ----
    zero_f32<<<(int)((nFeat + 255) / 256), 256, 0, stream>>>(agg, nFeat);
    scatter_add<<<scatterBlocks, 256, 0, stream>>>(src, tgt, x, agg, NEDGES);
    sage_gemm<<<gemmBlocks, 256, 0, stream>>>(agg, deg, x, Wsw1, b1, h, 1);

    // ---- layer 2 ----
    zero_f32<<<(int)((nFeat + 255) / 256), 256, 0, stream>>>(agg, nFeat);
    scatter_add<<<scatterBlocks, 256, 0, stream>>>(src, tgt, h, agg, NEDGES);
    sage_gemm<<<gemmBlocks, 256, 0, stream>>>(agg, deg, h, Wsw2, b2, out, 0);
}
